// CodeBook_27917287424806
// MI455X (gfx1250) — compile-verified, matrix operand order checked by reading, not run on hardware
//
#include <hip/hip_runtime.h>
#include <hip/hip_bf16.h>

typedef __attribute__((ext_vector_type(16))) __bf16 v16bf;
typedef __attribute__((ext_vector_type(8)))  __bf16 v8bf;
typedef __attribute__((ext_vector_type(8)))  float  v8f;

#define NN    20000
#define DINC  1280
#define HC    256
#define KCB   4096
#define NEC   320000
#define BN_EPS 1e-5f

// GEMM block tile
#define BM 128
#define BNT 64
#define BK 32
#define LDK 40   // LDS row stride in shorts (80B, 16B aligned)

union FragU {
  v16bf v;
  v8bf  h[2];
  unsigned int u[8];
};

__device__ __forceinline__ unsigned short f2bfu(float x) {
  return (unsigned short)(__float_as_uint(x) >> 16);
}
__device__ __forceinline__ unsigned int pack2bf(float lo, float hi) {
  return (__float_as_uint(hi) & 0xFFFF0000u) | (__float_as_uint(lo) >> 16);
}

// ---------------------------------------------------------------------------
// Tiled GEMM: C[M,Nn] = A[M,Kd]*B[Kd,Nn] (+bias)(relu). fp32 in/out, bf16 WMMA.
// Block = 256 thr = 8 waves; tile 128x64; wave tile 32x32 (2x2 WMMA frags).
// LDS: A as [m][k], B transposed as [n][k] -> fragment halves are contiguous
// 16B runs, loaded as ds_load_b128.
// ---------------------------------------------------------------------------
__global__ __launch_bounds__(256)
void gemm_bf16_wmma(const float* __restrict__ A, const float* __restrict__ B,
                    const float* __restrict__ bias, float* __restrict__ C,
                    int M, int Kd, int Nn, int relu)
{
  __shared__ __align__(16) unsigned short sA[BM][LDK];
  __shared__ __align__(16) unsigned short sBt[BNT][LDK];
  const int tid  = threadIdx.x;
  const int wave = tid >> 5;
  const int lane = tid & 31;
  const int half = lane >> 4;
  const int li   = lane & 15;
  const int bm = blockIdx.y * BM;
  const int bn = blockIdx.x * BNT;
  const int wm = (wave >> 1) << 5;   // 0,32,64,96
  const int wn = (wave & 1) << 5;    // 0,32

  v8f acc[2][2] = {};

  for (int k0 = 0; k0 < Kd; k0 += BK) {
    // ---- stage A tile: BM x 32 floats as bf16 [m][k] ----
    for (int i = tid; i < BM * 8; i += 256) {       // float4 chunks
      int r = i >> 3, c4 = (i & 7) << 2;
      int gm = bm + r;
      float4 v = make_float4(0.f, 0.f, 0.f, 0.f);
      if (gm < M) {
        v = *reinterpret_cast<const float4*>(A + (size_t)gm * Kd + (k0 + c4));
        if (k0 + BK < Kd)
          __builtin_prefetch(A + (size_t)gm * Kd + (k0 + BK + c4), 0, 3);
      }
      ushort4 h;
      h.x = f2bfu(v.x); h.y = f2bfu(v.y); h.z = f2bfu(v.z); h.w = f2bfu(v.w);
      *reinterpret_cast<ushort4*>(&sA[r][c4]) = h;
    }
    // ---- stage B tile: 32 x BNT floats, transposed to bf16 [n][k] ----
    for (int i = tid; i < (BK * BNT) / 4; i += 256) {  // 512 float4 chunks
      int r = i >> 4, c4 = (i & 15) << 2;              // r: k row, c4: n col
      float4 v = *reinterpret_cast<const float4*>(B + (size_t)(k0 + r) * Nn + (bn + c4));
      if (k0 + BK < Kd)
        __builtin_prefetch(B + (size_t)(k0 + BK + r) * Nn + (bn + c4), 0, 3);
      sBt[c4 + 0][r] = f2bfu(v.x);
      sBt[c4 + 1][r] = f2bfu(v.y);
      sBt[c4 + 2][r] = f2bfu(v.z);
      sBt[c4 + 3][r] = f2bfu(v.w);
    }
    __syncthreads();

    FragU af[2], bf[2];
    const int kb = half * 8;
#pragma unroll
    for (int t = 0; t < 2; ++t) {
      const int am = wm + t * 16 + li;
      af[t].h[0] = *reinterpret_cast<const v8bf*>(&sA[am][kb]);
      af[t].h[1] = *reinterpret_cast<const v8bf*>(&sA[am][kb + 16]);
      const int bc = wn + t * 16 + li;
      bf[t].h[0] = *reinterpret_cast<const v8bf*>(&sBt[bc][kb]);
      bf[t].h[1] = *reinterpret_cast<const v8bf*>(&sBt[bc][kb + 16]);
    }
#pragma unroll
    for (int ti = 0; ti < 2; ++ti)
#pragma unroll
      for (int tj = 0; tj < 2; ++tj)
        acc[ti][tj] = __builtin_amdgcn_wmma_f32_16x16x32_bf16(
            false, af[ti].v, false, bf[tj].v, (short)0, acc[ti][tj], false, false);
    __syncthreads();
  }

  // ---- epilogue: C slot j -> row = 16*ti + 8*half + j, col = 16*tj + li ----
#pragma unroll
  for (int ti = 0; ti < 2; ++ti) {
#pragma unroll
    for (int tj = 0; tj < 2; ++tj) {
      const int col = bn + wn + tj * 16 + li;
      const float bv = bias ? bias[col] : 0.0f;
#pragma unroll
      for (int j = 0; j < 8; ++j) {
        int row = bm + wm + ti * 16 + half * 8 + j;
        if (row < M) {
          float v = acc[ti][tj][j] + bv;
          if (relu) v = fmaxf(v, 0.0f);
          C[(size_t)row * Nn + col] = v;
        }
      }
    }
  }
}

// ---------------------------------------------------------------------------
// Fused VQ argmax: 16 rows/wave, scan 4096 normalized codes with WMMA dots,
// track argmax (== argmin L2 on normalized vectors, first-index tie-break).
// ---------------------------------------------------------------------------
__global__ __launch_bounds__(32)
void vq_argmax_wmma(const float* __restrict__ zn, const float* __restrict__ ewn,
                    int* __restrict__ idx)
{
  const int row0 = blockIdx.x * 16;
  const int lane = threadIdx.x;
  const int half = lane >> 4;
  const int li   = lane & 15;

  FragU afr[8];
  const float* arow = zn + (size_t)(row0 + li) * HC;
#pragma unroll
  for (int kk = 0; kk < 8; ++kk) {
    const int kbase = kk * 32 + half * 8;
    float4 p0 = *reinterpret_cast<const float4*>(arow + kbase);
    float4 p1 = *reinterpret_cast<const float4*>(arow + kbase + 4);
    float4 p2 = *reinterpret_cast<const float4*>(arow + kbase + 16);
    float4 p3 = *reinterpret_cast<const float4*>(arow + kbase + 20);
    afr[kk].u[0] = pack2bf(p0.x, p0.y); afr[kk].u[1] = pack2bf(p0.z, p0.w);
    afr[kk].u[2] = pack2bf(p1.x, p1.y); afr[kk].u[3] = pack2bf(p1.z, p1.w);
    afr[kk].u[4] = pack2bf(p2.x, p2.y); afr[kk].u[5] = pack2bf(p2.z, p2.w);
    afr[kk].u[6] = pack2bf(p3.x, p3.y); afr[kk].u[7] = pack2bf(p3.z, p3.w);
  }
  float bestv[8];
  int   besti[8];
#pragma unroll
  for (int j = 0; j < 8; ++j) { bestv[j] = -1e30f; besti[j] = 0; }

  for (int n0 = 0; n0 < KCB; n0 += 16) {
    v8f c = {};
    const float* brow = ewn + (size_t)(n0 + li) * HC;
#pragma unroll
    for (int kk = 0; kk < 8; ++kk) {
      const int kbase = kk * 32 + half * 8;
      float4 p0 = *reinterpret_cast<const float4*>(brow + kbase);
      float4 p1 = *reinterpret_cast<const float4*>(brow + kbase + 4);
      float4 p2 = *reinterpret_cast<const float4*>(brow + kbase + 16);
      float4 p3 = *reinterpret_cast<const float4*>(brow + kbase + 20);
      FragU bfr;
      bfr.u[0] = pack2bf(p0.x, p0.y); bfr.u[1] = pack2bf(p0.z, p0.w);
      bfr.u[2] = pack2bf(p1.x, p1.y); bfr.u[3] = pack2bf(p1.z, p1.w);
      bfr.u[4] = pack2bf(p2.x, p2.y); bfr.u[5] = pack2bf(p2.z, p2.w);
      bfr.u[6] = pack2bf(p3.x, p3.y); bfr.u[7] = pack2bf(p3.z, p3.w);
      c = __builtin_amdgcn_wmma_f32_16x16x32_bf16(false, afr[kk].v, false, bfr.v,
                                                  (short)0, c, false, false);
    }
    const int col = n0 + li;
#pragma unroll
    for (int j = 0; j < 8; ++j) {
      if (c[j] > bestv[j]) { bestv[j] = c[j]; besti[j] = col; }
    }
  }
#pragma unroll
  for (int j = 0; j < 8; ++j) {
    float v = bestv[j];
    int   bi = besti[j];
#pragma unroll
    for (int o = 8; o >= 1; o >>= 1) {
      float ov = __shfl_xor(v, o, 32);
      int   oi = __shfl_xor(bi, o, 32);
      if (ov > v || (ov == v && oi < bi)) { v = ov; bi = oi; }
    }
    if (li == 0) idx[row0 + half * 8 + j] = bi;
  }
}

// ---------------------------------------------------------------------------
// Graph / elementwise helpers
// ---------------------------------------------------------------------------
__global__ void zero_kernel(float* __restrict__ p, size_t n) {
  size_t t = (size_t)blockIdx.x * blockDim.x + threadIdx.x;
  if (t < n) p[t] = 0.0f;
}

__global__ void degree_kernel(const int* __restrict__ src, const int* __restrict__ dst,
                              float* __restrict__ dego, float* __restrict__ degi) {
  int e = blockIdx.x * blockDim.x + threadIdx.x;
  if (e >= NEC) return;
  atomicAdd(&dego[src[e]], 1.0f);
  atomicAdd(&degi[dst[e]], 1.0f);
}

__global__ void invsqrt_kernel(float* __restrict__ c) {
  int i = blockIdx.x * blockDim.x + threadIdx.x;
  if (i >= NN) return;
  float d = c[i];
  c[i] = (d > 0.0f) ? 1.0f / sqrtf(fmaxf(d, 1.0f)) : 0.0f;
}

// agg[dst] += hw[src] * c_src[src]  (edge-parallel, float4 chunks, L2-resident)
__global__ __launch_bounds__(256)
void scatter_kernel(const int* __restrict__ src, const int* __restrict__ dst,
                    const float* __restrict__ cs, const float* __restrict__ hw,
                    float* __restrict__ agg) {
  size_t t = (size_t)blockIdx.x * blockDim.x + threadIdx.x;
  size_t e = t >> 6;
  if (e >= (size_t)NEC) return;
  int c4 = (int)(t & 63) << 2;
  int s = src[e], d = dst[e];
  float w = cs[s];
  const float4 v = *reinterpret_cast<const float4*>(hw + (size_t)s * HC + c4);
  float* o = agg + (size_t)d * HC + c4;
  atomicAdd(o + 0, v.x * w);
  atomicAdd(o + 1, v.y * w);
  atomicAdd(o + 2, v.z * w);
  atomicAdd(o + 3, v.w * w);
}

// acc += agg * c_dst[row] + b[col]
__global__ void scale_add_kernel(float* __restrict__ acc, const float* __restrict__ agg,
                                 const float* __restrict__ cd, const float* __restrict__ b) {
  size_t t = (size_t)blockIdx.x * blockDim.x + threadIdx.x;
  if (t >= (size_t)NN * HC) return;
  int i = (int)(t >> 8);
  int c = (int)(t & 255);
  acc[t] += agg[t] * cd[i] + b[c];
}

__global__ __launch_bounds__(256)
void bn_stats_kernel(const float* __restrict__ x, float* __restrict__ sum,
                     float* __restrict__ sumsq, int rows_per_block) {
  int c = threadIdx.x;  // blockDim == HC
  int r0 = blockIdx.x * rows_per_block;
  int r1 = r0 + rows_per_block; if (r1 > NN) r1 = NN;
  float s = 0.0f, q = 0.0f;
  for (int r = r0; r < r1; ++r) {
    float v = x[(size_t)r * HC + c];
    s += v; q += v * v;
  }
  atomicAdd(&sum[c], s);
  atomicAdd(&sumsq[c], q);
}

__global__ void bn_apply_kernel(const float* __restrict__ x, const float* __restrict__ sum,
                                const float* __restrict__ sumsq, const float* __restrict__ g,
                                const float* __restrict__ beta, float* __restrict__ y) {
  size_t t = (size_t)blockIdx.x * blockDim.x + threadIdx.x;
  if (t >= (size_t)NN * HC) return;
  int c = (int)(t & 255);
  float mu  = sum[c] * (1.0f / NN);
  float var = sumsq[c] * (1.0f / NN) - mu * mu;
  y[t] = (x[t] - mu) * (1.0f / sqrtf(var + BN_EPS)) * g[c] + beta[c];
}

// one wave per row: y = x / max(||x||, 1e-12)
__global__ __launch_bounds__(256)
void rownorm_kernel(const float* __restrict__ x, float* __restrict__ y, int M, int C) {
  int wave = threadIdx.x >> 5, lane = threadIdx.x & 31;
  int row = blockIdx.x * 8 + wave;
  if (row >= M) return;
  const float* xr = x + (size_t)row * C;
  float s = 0.0f;
  for (int c = lane; c < C; c += 32) { float v = xr[c]; s += v * v; }
#pragma unroll
  for (int o = 16; o >= 1; o >>= 1) s += __shfl_xor(s, o, 32);
  float inv = 1.0f / fmaxf(sqrtf(s), 1e-12f);
  float* yr = y + (size_t)row * C;
  for (int c = lane; c < C; c += 32) yr[c] = xr[c] * inv;
}

// e = ewn[idx], e_m = mask[idx] ? 0 : e
__global__ void vq_gather_kernel(const int* __restrict__ idx, const float* __restrict__ ewn,
                                 const unsigned char* __restrict__ mk,
                                 float* __restrict__ e, float* __restrict__ em) {
  size_t t = (size_t)blockIdx.x * blockDim.x + threadIdx.x;
  if (t >= (size_t)NN * HC) return;
  int i = (int)(t >> 8), c = (int)(t & 255);
  int id = idx[i];
  float v = ewn[(size_t)id * HC + c];
  e[t] = v;
  em[t] = mk[id] ? 0.0f : v;
}

__global__ __launch_bounds__(256)
void sqdiff_kernel(const float* __restrict__ a, const float* __restrict__ b,
                   float* __restrict__ out, size_t n) {
  __shared__ float sh[256];
  size_t t = (size_t)blockIdx.x * blockDim.x + threadIdx.x;
  size_t stride = (size_t)gridDim.x * blockDim.x;
  float s = 0.0f;
  for (size_t i = t; i < n; i += stride) { float d = a[i] - b[i]; s += d * d; }
  sh[threadIdx.x] = s;
  __syncthreads();
  for (int st = 128; st >= 1; st >>= 1) {
    if ((int)threadIdx.x < st) sh[threadIdx.x] += sh[threadIdx.x + st];
    __syncthreads();
  }
  if (threadIdx.x == 0) atomicAdd(out, sh[0]);
}

// per-row SCE loss on masked rows: (1 - cos(xr, x))^2
__global__ __launch_bounds__(256)
void sce_loss_kernel(const float* __restrict__ xr, const float* __restrict__ x,
                     const int* __restrict__ idx, const unsigned char* __restrict__ mk,
                     float* __restrict__ msum, float* __restrict__ mcnt) {
  int wave = threadIdx.x >> 5, lane = threadIdx.x & 31;
  int row = blockIdx.x * 8 + wave;
  if (row >= NN) return;
  const float* a = xr + (size_t)row * DINC;
  const float* b = x  + (size_t)row * DINC;
  float dot = 0.0f, na = 0.0f, nb = 0.0f;
  for (int c = lane; c < DINC; c += 32) {
    float u = a[c], v = b[c];
    dot += u * v; na += u * u; nb += v * v;
  }
#pragma unroll
  for (int o = 16; o >= 1; o >>= 1) {
    dot += __shfl_xor(dot, o, 32);
    na  += __shfl_xor(na,  o, 32);
    nb  += __shfl_xor(nb,  o, 32);
  }
  if (lane == 0 && mk[idx[row]]) {
    float d = dot / (fmaxf(sqrtf(na), 1e-12f) * fmaxf(sqrtf(nb), 1e-12f));
    float per = 1.0f - d;
    per = per * per;   // SCE = 2.0
    atomicAdd(msum, per);
    atomicAdd(mcnt, 1.0f);
  }
}

__global__ void finalize_kernel(const float* __restrict__ scal, float* __restrict__ out3) {
  if (threadIdx.x == 0 && blockIdx.x == 0) {
    out3[0] = scal[0] * (1.25f / ((float)NN * (float)HC));   // e_q_loss = (1+CC)*mean
    out3[1] = scal[1] * (1.0f / ((float)NN * (float)DINC));  // recon_loss
    out3[2] = scal[2] / (scal[3] + 1e-12f);                  // mask_loss
  }
}

// ---------------------------------------------------------------------------
// Host orchestration
// ---------------------------------------------------------------------------
static inline void launch_gemm(const float* A, const float* B, const float* bias, float* C,
                               int M, int Kd, int Nn, int relu, hipStream_t s) {
  dim3 g((Nn + BNT - 1) / BNT, (M + BM - 1) / BM);
  gemm_bf16_wmma<<<g, 256, 0, s>>>(A, B, bias, C, M, Kd, Nn, relu);
}

extern "C" void kernel_launch(void* const* d_in, const int* in_sizes, int n_in,
                              void* d_out, int out_size, void* d_ws, size_t ws_size,
                              hipStream_t stream) {
  (void)in_sizes; (void)n_in; (void)out_size; (void)ws_size;
  auto F = [&](int i) { return (const float*)d_in[i]; };

  const float* x = F(0);
  const int* edges[3] = { (const int*)d_in[1], (const int*)d_in[2], (const int*)d_in[3] };
  const unsigned char* maskK = (const unsigned char*)d_in[4];

  const float *enc_w[3][3], *enc_b[3][3], *enc_fc_w[3], *enc_fc_b[3], *enc_g[3], *enc_beta[3];
  const float *dec_w[3][3], *dec_b[3][3], *dec_fc_w[3], *dec_fc_b[3], *dec_g[2], *dec_beta[2];
  int p = 5;
  for (int l = 0; l < 3; ++l) for (int r = 0; r < 3; ++r) enc_w[l][r] = F(p++);
  for (int l = 0; l < 3; ++l) for (int r = 0; r < 3; ++r) enc_b[l][r] = F(p++);
  for (int l = 0; l < 3; ++l) enc_fc_w[l] = F(p++);
  for (int l = 0; l < 3; ++l) enc_fc_b[l] = F(p++);
  for (int l = 0; l < 3; ++l) enc_g[l]    = F(p++);
  for (int l = 0; l < 3; ++l) enc_beta[l] = F(p++);
  for (int l = 0; l < 3; ++l) for (int r = 0; r < 3; ++r) dec_w[l][r] = F(p++);
  for (int l = 0; l < 3; ++l) for (int r = 0; r < 3; ++r) dec_b[l][r] = F(p++);
  for (int l = 0; l < 3; ++l) dec_fc_w[l] = F(p++);
  for (int l = 0; l < 3; ++l) dec_fc_b[l] = F(p++);
  for (int l = 0; l < 2; ++l) dec_g[l]    = F(p++);
  for (int l = 0; l < 2; ++l) dec_beta[l] = F(p++);
  const float* emb = F(p++);

  // ---- workspace carve-out (~251 MB of fp32) ----
  const size_t NH = (size_t)NN * HC;
  float* ws = (float*)d_ws;
  size_t off = 0;
  auto alloc = [&](size_t n) { float* q = ws + off; off += n; return q; };
  float *cs[3], *cd[3];
  for (int r = 0; r < 3; ++r) { cs[r] = alloc(NN); cd[r] = alloc(NN); }
  float* tmp   = alloc(NH);
  float* agg   = alloc(NH);
  float* acc   = alloc(NH);
  float* h_a   = alloc(NH);
  float* h_b   = alloc(NH);
  float* zn    = alloc(NH);
  float* e_buf = alloc(NH);
  float* ewn   = alloc((size_t)KCB * HC);
  int*   idx   = (int*)alloc(NN);
  float* xr    = alloc((size_t)NN * DINC);
  float* bnsum = alloc(HC);
  float* bnsq  = alloc(HC);
  float* scal  = alloc(8);

  float* z_out  = (float*)d_out;
  float* em_out = (float*)d_out + NH;
  float* out3   = (float*)d_out + 2 * NH;

  auto gz = [](size_t n) { return dim3((unsigned)((n + 255) / 256)); };
  const dim3 gNH   = gz(NH);
  const dim3 gScat = gz((size_t)NEC * 64);

  // ---- per-relation degree normalizers (once per call) ----
  for (int r = 0; r < 3; ++r) {
    zero_kernel<<<gz(NN), 256, 0, stream>>>(cs[r], NN);
    zero_kernel<<<gz(NN), 256, 0, stream>>>(cd[r], NN);
    degree_kernel<<<gz(NEC), 256, 0, stream>>>(edges[r], edges[r] + NEC, cs[r], cd[r]);
    invsqrt_kernel<<<gz(NN), 256, 0, stream>>>(cs[r]);
    invsqrt_kernel<<<gz(NN), 256, 0, stream>>>(cd[r]);
  }

  auto hetero = [&](const float* hin, int Kd, const float* (*w)[3], const float* (*b)[3], int l) {
    zero_kernel<<<gNH, 256, 0, stream>>>(acc, NH);
    for (int r = 0; r < 3; ++r) {
      launch_gemm(hin, w[l][r], nullptr, tmp, NN, Kd, HC, 0, stream);
      zero_kernel<<<gNH, 256, 0, stream>>>(agg, NH);
      scatter_kernel<<<gScat, 256, 0, stream>>>(edges[r], edges[r] + NEC, cs[r], tmp, agg);
      scale_add_kernel<<<gNH, 256, 0, stream>>>(acc, agg, cd[r], b[l][r]);
    }
  };
  auto batchnorm = [&](const float* in, const float* g, const float* beta, float* out) {
    zero_kernel<<<gz(HC), 256, 0, stream>>>(bnsum, HC);
    zero_kernel<<<gz(HC), 256, 0, stream>>>(bnsq, HC);
    bn_stats_kernel<<<200, 256, 0, stream>>>(in, bnsum, bnsq, 100);
    bn_apply_kernel<<<gNH, 256, 0, stream>>>(in, bnsum, bnsq, g, beta, out);
  };

  // ---- encoder: x -> h_a -> h_b -> z_out ----
  {
    const float* hin = x;
    float* outs[3] = { h_a, h_b, z_out };
    for (int l = 0; l < 3; ++l) {
      int Kd = (l == 0) ? DINC : HC;
      hetero(hin, Kd, enc_w, enc_b, l);
      launch_gemm(acc, enc_fc_w[l], enc_fc_b[l], tmp, NN, HC, HC, 1, stream);
      batchnorm(tmp, enc_g[l], enc_beta[l], outs[l]);
      hin = outs[l];
    }
  }

  // ---- vector quantizer ----
  rownorm_kernel<<<(NN + 7) / 8, 256, 0, stream>>>(z_out, zn, NN, HC);
  rownorm_kernel<<<(KCB + 7) / 8, 256, 0, stream>>>(emb, ewn, KCB, HC);
  vq_argmax_wmma<<<NN / 16, 32, 0, stream>>>(zn, ewn, idx);
  vq_gather_kernel<<<gNH, 256, 0, stream>>>(idx, ewn, maskK, e_buf, em_out);
  zero_kernel<<<1, 256, 0, stream>>>(scal, 8);
  sqdiff_kernel<<<1024, 256, 0, stream>>>(e_buf, zn, &scal[0], NH);

  // ---- decoder (run twice: e_buf -> recon loss; em_out -> masked SCE loss) ----
  auto run_decoder = [&](const float* din) {
    const float* hin = din;
    for (int l = 0; l < 3; ++l) {
      hetero(hin, HC, dec_w, dec_b, l);
      if (l < 2) {
        launch_gemm(acc, dec_fc_w[l], dec_fc_b[l], tmp, NN, HC, HC, 1, stream);
        float* hout = (l == 0) ? h_a : h_b;
        batchnorm(tmp, dec_g[l], dec_beta[l], hout);
        hin = hout;
      } else {
        launch_gemm(acc, dec_fc_w[2], dec_fc_b[2], xr, NN, HC, DINC, 0, stream);
      }
    }
  };

  run_decoder(e_buf);
  sqdiff_kernel<<<2048, 256, 0, stream>>>(xr, x, &scal[1], (size_t)NN * DINC);

  run_decoder(em_out);
  sce_loss_kernel<<<(NN + 7) / 8, 256, 0, stream>>>(xr, x, idx, maskK, &scal[2], &scal[3]);

  finalize_kernel<<<1, 1, 0, stream>>>(scal, out3);
}
